// SelfAttention_47940424958205
// MI455X (gfx1250) — matrix operand order repaired and checked by reading, not compiled, not run on hardware
//
#include <hip/hip_runtime.h>
#include <hip/hip_bf16.h>

typedef __bf16 bf16_t;
typedef __attribute__((ext_vector_type(16))) __bf16 v16bf;
typedef __attribute__((ext_vector_type(8)))  __bf16 v8bf;
typedef __attribute__((ext_vector_type(8)))  float  v8f;
typedef int v4i __attribute__((vector_size(16)));

#define EMBED 1024
#define INNER 512
#define NQKV  1536
#define NHEAD 8
#define HDIM  64
#define SEQ   4096
#define BATCH 2
#define MTOT  (BATCH*SEQ)   // 8192

// Padded LDS leading dims (elements) to spread the 64 banks.
#define BS_LD 40   // GEMM B tile: 64 rows x 32 cols (+8 pad)
#define KS_LD 72   // K/V tiles:   64 rows x 64 cols (+8 pad)
#define PB_LD 72   // P tile:      16 rows x 64 cols (+8 pad)

// ---------------------------------------------------------------------------
// CDNA5 async global->LDS path (ASYNCcnt), with safe fallback.
// Probe-confirmed signature shape: (v4i addrspace(1)*, v4i addrspace(3)*, Ii, Ii)
// ---------------------------------------------------------------------------
#if defined(__AMDGCN__) && __has_builtin(__builtin_amdgcn_global_load_async_to_lds_b128) && __has_builtin(__builtin_amdgcn_s_wait_asynccnt)
#define USE_ASYNC 1
#else
#define USE_ASYNC 0
#endif

__device__ __forceinline__ void async_copy16(bf16_t* l, const bf16_t* g) {
#if USE_ASYNC
  __builtin_amdgcn_global_load_async_to_lds_b128(
      (__attribute__((address_space(1))) v4i*)(bf16_t*)g,
      (__attribute__((address_space(3))) v4i*)l,
      0, 0);
#else
  *(v8bf*)l = *(const v8bf*)g;
#endif
}

__device__ __forceinline__ void async_wait() {
#if USE_ASYNC
  __builtin_amdgcn_s_wait_asynccnt(0);
#endif
}

// ---------------------------------------------------------------------------
// WMMA helpers (v_wmma_f32_16x16x32_bf16): D = A(16x32) * B(32x16) + C
// ---------------------------------------------------------------------------
__device__ __forceinline__ v8f wmma_bf16(v16bf a, v16bf b, v8f c) {
  return __builtin_amdgcn_wmma_f32_16x16x32_bf16(
      false, a, false, b, (short)0, c, false, false);
}

// A-fragment 16(M) x 32(K), row-major source, leading dim ld (elements).
__device__ __forceinline__ v16bf load_a_frag(const bf16_t* p, int ld, int lane) {
  int m = lane & 15, kh = lane >> 4;
  const bf16_t* row = p + (size_t)m * ld;
  union { v16bf v; v8bf h[2]; } u;
  u.h[0] = *(const v8bf*)(row + 8 * kh);
  u.h[1] = *(const v8bf*)(row + 16 + 8 * kh);
  return u.v;
}

// B-fragment 32(K) x 16(N) from the TRANSPOSED operand Bt[N][K], ld elements.
__device__ __forceinline__ v16bf load_b_frag(const bf16_t* pT, int ld, int lane) {
  int n = lane & 15, kh = lane >> 4;
  const bf16_t* col = pT + (size_t)n * ld;
  union { v16bf v; v8bf h[2]; } u;
  u.h[0] = *(const v8bf*)(col + 8 * kh);
  u.h[1] = *(const v8bf*)(col + 16 + 8 * kh);
  return u.v;
}

// ---------------------------------------------------------------------------
// Conversion / layout kernels
// ---------------------------------------------------------------------------
__global__ void cvt_f32_to_bf16(const float* __restrict__ src,
                                bf16_t* __restrict__ dst, int n) {
  int i = blockIdx.x * blockDim.x + threadIdx.x;
  if (i < n) dst[i] = (bf16_t)src[i];
}

__global__ void transpose_cvt(const float* __restrict__ src,
                              bf16_t* __restrict__ dst, int rows, int cols) {
  int i = blockIdx.x * blockDim.x + threadIdx.x;
  if (i < rows * cols) {
    int r = i / cols, c = i - r * cols;
    dst[(size_t)c * rows + r] = (bf16_t)src[i];
  }
}

// ---------------------------------------------------------------------------
// Shared staged-GEMM core. Block = 256 threads = 8 waves.
// Block tile 256(M) x 64(N); wave tile 32(M) x 64(N); K step 32.
// B tile (64 x 32) staged via async copies into double-buffered LDS.
// ---------------------------------------------------------------------------
__device__ __forceinline__ void stage_b(bf16_t* bs, const bf16_t* BtT, int ldb,
                                        int k0, int tid) {
  int r = tid >> 2, col = (tid & 3) * 8;     // 256 threads -> 64 rows x 4 chunks
  async_copy16(bs + r * BS_LD + col, BtT + (size_t)r * ldb + k0 + col);
}

template <int KTOT>
__device__ __forceinline__ void gemm_core(const bf16_t* __restrict__ Arow0,
                                          const bf16_t* __restrict__ Arow1,
                                          const bf16_t* __restrict__ BtT, // +n0*KTOT
                                          bf16_t* bs,                     // [2][64*BS_LD]
                                          int lane, int tid, v8f acc[2][4]) {
  stage_b(bs, BtT, KTOT, 0, tid);
  v16bf a0 = load_a_frag(Arow0, KTOT, lane);
  v16bf a1 = load_a_frag(Arow1, KTOT, lane);
  int buf = 0;
  for (int k0 = 0; k0 < KTOT; k0 += 32, buf ^= 1) {
    async_wait();
    __syncthreads();
    int kn = (k0 + 32 < KTOT) ? (k0 + 32) : k0;   // clamped prefetch
    stage_b(bs + (buf ^ 1) * (64 * BS_LD), BtT, KTOT, kn, tid);
    v16bf na0 = load_a_frag(Arow0 + kn, KTOT, lane);
    v16bf na1 = load_a_frag(Arow1 + kn, KTOT, lane);
    const bf16_t* bp = bs + buf * (64 * BS_LD);
#pragma unroll
    for (int nt = 0; nt < 4; ++nt) {
      v16bf b = load_b_frag(bp + 16 * nt * BS_LD, BS_LD, lane);
      acc[0][nt] = wmma_bf16(a0, b, acc[0][nt]);
      acc[1][nt] = wmma_bf16(a1, b, acc[1][nt]);
    }
    __syncthreads();
    a0 = na0; a1 = na1;
  }
}

// ---------------------------------------------------------------------------
// Kernel 1: fused QKV projection -> Qh (pre-scaled 1/8), Kh, Vt (transposed).
// ---------------------------------------------------------------------------
__global__ void qkv_gemm(const bf16_t* __restrict__ X,
                         const bf16_t* __restrict__ WqkvT,   // [1536][1024]
                         const float*  __restrict__ bias,    // [1536]
                         bf16_t* __restrict__ Qh, bf16_t* __restrict__ Kh,
                         bf16_t* __restrict__ Vt) {
  __shared__ __align__(16) bf16_t bs[2][64 * BS_LD];
  const int tid = threadIdx.x, lane = tid & 31, wave = tid >> 5;
  const int m0 = blockIdx.y * 256 + wave * 32;
  const int n0 = blockIdx.x * 64;

  v8f acc[2][4] = {};
  gemm_core<EMBED>(X + (size_t)m0 * EMBED, X + (size_t)(m0 + 16) * EMBED,
                   WqkvT + (size_t)n0 * EMBED, bs[0], lane, tid, acc);

  const int g = lane >> 4, nlo = lane & 15;
#pragma unroll
  for (int mt = 0; mt < 2; ++mt)
#pragma unroll
    for (int nt = 0; nt < 4; ++nt) {
      int o = n0 + 16 * nt + nlo;
      float bv = bias[o];
#pragma unroll
      for (int j = 0; j < 8; ++j) {
        int token = m0 + 16 * mt + 8 * g + j;
        int bb = token >> 12, nseq = token & (SEQ - 1);
        float v = acc[mt][nt][j] + bv;
        if (o < INNER) {                      // Q (fold in softmax scale 1/8)
          int h = o >> 6, d = o & 63;
          Qh[(((size_t)(bb * NHEAD + h)) * SEQ + nseq) * HDIM + d] = (bf16_t)(v * 0.125f);
        } else if (o < 2 * INNER) {           // K
          int oo = o - INNER; int h = oo >> 6, d = oo & 63;
          Kh[(((size_t)(bb * NHEAD + h)) * SEQ + nseq) * HDIM + d] = (bf16_t)v;
        } else {                              // V, transposed [d][n]
          int oo = o - 2 * INNER; int h = oo >> 6, d = oo & 63;
          Vt[(((size_t)(bb * NHEAD + h)) * HDIM + d) * SEQ + nseq] = (bf16_t)v;
        }
      }
    }
}

// ---------------------------------------------------------------------------
// Kernel 2: flash attention. Block = 128 threads = 4 waves, 64 queries/block.
// 64-key blocks (K and V tiles) staged into double-buffered LDS via async
// copies shared by all 4 waves; online softmax; P re-enters A-layout via LDS.
// ---------------------------------------------------------------------------
__device__ __forceinline__ void stage_kv(bf16_t* ks, bf16_t* vs,
                                         const bf16_t* Kp, const bf16_t* Vp,
                                         int kb, int tid) {
#pragma unroll
  for (int i = 0; i < 4; ++i) {              // K rows contiguous in memory
    int c = i * 128 + tid;                   // 512 chunks of 16B
    int r = c >> 3, col = (c & 7) * 8;
    async_copy16(ks + r * KS_LD + col, Kp + (size_t)(kb + r) * HDIM + col);
  }
#pragma unroll
  for (int i = 0; i < 4; ++i) {              // V rows are d, cols are keys
    int c = i * 128 + tid;
    int r = c >> 3, col = (c & 7) * 8;
    async_copy16(vs + r * KS_LD + col, Vp + (size_t)r * SEQ + kb + col);
  }
}

__global__ void flash_attn(const bf16_t* __restrict__ Qh,
                           const bf16_t* __restrict__ Kh,
                           const bf16_t* __restrict__ Vt,
                           bf16_t* __restrict__ Attn /* [8192][512] */) {
  const int tid = threadIdx.x;
  const int lane = tid & 31;
  const int wave = tid >> 5;          // 0..3
  const int bh   = blockIdx.x >> 6;   // b*8+h
  const int qblk = blockIdx.x & 63;
  const int q0   = qblk * 64 + wave * 16;

  const bf16_t* Qp = Qh + ((size_t)bh * SEQ + q0) * HDIM;
  const bf16_t* Kp = Kh + (size_t)bh * SEQ * HDIM;
  const bf16_t* Vp = Vt + (size_t)bh * HDIM * SEQ;

  __shared__ __align__(16) bf16_t ks[2][64 * KS_LD];
  __shared__ __align__(16) bf16_t vs[2][64 * KS_LD];
  __shared__ __align__(16) bf16_t pbuf[4][16 * PB_LD];
  bf16_t* pb = pbuf[wave];

  const v16bf aq0 = load_a_frag(Qp, HDIM, lane);       // d 0..31
  const v16bf aq1 = load_a_frag(Qp + 32, HDIM, lane);  // d 32..63

  v8f O[4] = {};
  float mrow[8], lrow[8];
#pragma unroll
  for (int j = 0; j < 8; ++j) { mrow[j] = -__builtin_inff(); lrow[j] = 0.0f; }

  const int g = lane >> 4, nlo = lane & 15;

  stage_kv(ks[0], vs[0], Kp, Vp, 0, tid);
  int buf = 0;
  for (int kb = 0; kb < SEQ; kb += 64, buf ^= 1) {
    async_wait();
    __syncthreads();
    int kn = (kb + 64 < SEQ) ? (kb + 64) : kb;
    stage_kv(ks[buf ^ 1], vs[buf ^ 1], Kp, Vp, kn, tid);
    const bf16_t* kbase = ks[buf];
    const bf16_t* vbase = vs[buf];

    // ---- S = (Q/8) K^T : four 16-key subtiles, chained over d ----
    v8f S[4];
#pragma unroll
    for (int t = 0; t < 4; ++t) {
      v16bf b0 = load_b_frag(kbase + (size_t)16 * t * KS_LD, KS_LD, lane);
      v16bf b1 = load_b_frag(kbase + (size_t)16 * t * KS_LD + 32, KS_LD, lane);
      v8f c = {};
      c = wmma_bf16(aq0, b0, c);
      c = wmma_bf16(aq1, b1, c);
      S[t] = c;
    }

    // ---- online softmax (rows live in 16-lane half-waves) ----
#pragma unroll
    for (int j = 0; j < 8; ++j) {
      float v = fmaxf(fmaxf(S[0][j], S[1][j]), fmaxf(S[2][j], S[3][j]));
#pragma unroll
      for (int off = 1; off < 16; off <<= 1) v = fmaxf(v, __shfl_xor(v, off, 16));
      float nm = fmaxf(mrow[j], v);
      float alpha = __expf(mrow[j] - nm);
      mrow[j] = nm;
      lrow[j] *= alpha;
#pragma unroll
      for (int nt = 0; nt < 4; ++nt) O[nt][j] *= alpha;
      float rs = 0.0f;
#pragma unroll
      for (int t = 0; t < 4; ++t) {
        float p = __expf(S[t][j] - nm);
        S[t][j] = p;
        rs += p;
      }
#pragma unroll
      for (int off = 1; off < 16; off <<= 1) rs += __shfl_xor(rs, off, 16);
      lrow[j] += rs;
    }

    // ---- P (C/D layout f32) -> LDS 16x64 bf16 -> A-fragments ----
#pragma unroll
    for (int t = 0; t < 4; ++t)
#pragma unroll
      for (int j = 0; j < 8; ++j)
        pb[(8 * g + j) * PB_LD + 16 * t + nlo] = (bf16_t)S[t][j];
    v16bf ap0 = load_a_frag(pb, PB_LD, lane);        // keys 0..31
    v16bf ap1 = load_a_frag(pb + 32, PB_LD, lane);   // keys 32..63

    // ---- O += P V ----
#pragma unroll
    for (int nt = 0; nt < 4; ++nt) {
      v16bf bv0 = load_b_frag(vbase + (size_t)16 * nt * KS_LD, KS_LD, lane);
      v16bf bv1 = load_b_frag(vbase + (size_t)16 * nt * KS_LD + 32, KS_LD, lane);
      O[nt] = wmma_bf16(ap0, bv0, O[nt]);
      O[nt] = wmma_bf16(ap1, bv1, O[nt]);
    }
    __syncthreads();
  }

  // ---- finalize: O /= l, store token-major bf16 ----
  const int h = bh & (NHEAD - 1), bb = bh >> 3;
#pragma unroll
  for (int j = 0; j < 8; ++j) {
    float inv = 1.0f / lrow[j];
    int token = bb * SEQ + q0 + 8 * g + j;
    bf16_t* dst = Attn + (size_t)token * INNER + h * HDIM;
#pragma unroll
    for (int nt = 0; nt < 4; ++nt)
      dst[16 * nt + nlo] = (bf16_t)(O[nt][j] * inv);
  }
}

// ---------------------------------------------------------------------------
// Kernel 3: output projection. out[8192 x 1024] = Attn[8192 x 512]*Wproj + b
// ---------------------------------------------------------------------------
__global__ void proj_gemm(const bf16_t* __restrict__ A,      // [8192][512]
                          const bf16_t* __restrict__ WprojT, // [1024][512]
                          const float*  __restrict__ bias,   // [1024]
                          float* __restrict__ out) {         // [8192][1024]
  __shared__ __align__(16) bf16_t bs[2][64 * BS_LD];
  const int tid = threadIdx.x, lane = tid & 31, wave = tid >> 5;
  const int m0 = blockIdx.y * 256 + wave * 32;
  const int n0 = blockIdx.x * 64;

  v8f acc[2][4] = {};
  gemm_core<INNER>(A + (size_t)m0 * INNER, A + (size_t)(m0 + 16) * INNER,
                   WprojT + (size_t)n0 * INNER, bs[0], lane, tid, acc);

  const int g = lane >> 4, nlo = lane & 15;
#pragma unroll
  for (int mt = 0; mt < 2; ++mt)
#pragma unroll
    for (int nt = 0; nt < 4; ++nt) {
      int o = n0 + 16 * nt + nlo;
      float bv = bias[o];
#pragma unroll
      for (int j = 0; j < 8; ++j) {
        int token = m0 + 16 * mt + 8 * g + j;
        out[(size_t)token * EMBED + o] = acc[mt][nt][j] + bv;
      }
    }
}

// ---------------------------------------------------------------------------
extern "C" void kernel_launch(void* const* d_in, const int* in_sizes, int n_in,
                              void* d_out, int out_size, void* d_ws, size_t ws_size,
                              hipStream_t stream) {
  (void)in_sizes; (void)n_in; (void)out_size; (void)ws_size;
  const float* X  = (const float*)d_in[0];  // [2,4096,1024]
  const float* Wq = (const float*)d_in[1];  // [1024,1536]
  const float* Bq = (const float*)d_in[2];  // [1536]
  const float* Wp = (const float*)d_in[3];  // [512,1024]
  const float* Bp = (const float*)d_in[4];  // [1024]
  float* out = (float*)d_out;

  char* ws = (char*)d_ws;
  bf16_t* Xbf = (bf16_t*)ws;  ws += (size_t)MTOT * EMBED * 2;
  bf16_t* WqT = (bf16_t*)ws;  ws += (size_t)NQKV * EMBED * 2;
  bf16_t* WpT = (bf16_t*)ws;  ws += (size_t)EMBED * INNER * 2;
  bf16_t* Qh  = (bf16_t*)ws;  ws += (size_t)BATCH * NHEAD * SEQ * HDIM * 2;
  bf16_t* Kh  = (bf16_t*)ws;  ws += (size_t)BATCH * NHEAD * SEQ * HDIM * 2;
  bf16_t* Vt  = (bf16_t*)ws;  ws += (size_t)BATCH * NHEAD * SEQ * HDIM * 2;
  bf16_t* At  = (bf16_t*)ws;  ws += (size_t)MTOT * INNER * 2;

  {
    int n = MTOT * EMBED;
    cvt_f32_to_bf16<<<(n + 255) / 256, 256, 0, stream>>>(X, Xbf, n);
  }
  {
    int n = EMBED * NQKV;
    transpose_cvt<<<(n + 255) / 256, 256, 0, stream>>>(Wq, WqT, EMBED, NQKV);
  }
  {
    int n = INNER * EMBED;
    transpose_cvt<<<(n + 255) / 256, 256, 0, stream>>>(Wp, WpT, INNER, EMBED);
  }

  qkv_gemm<<<dim3(NQKV / 64, MTOT / 256), 256, 0, stream>>>(Xbf, WqT, Bq, Qh, Kh, Vt);
  flash_attn<<<BATCH * NHEAD * (SEQ / 64), 128, 0, stream>>>(Qh, Kh, Vt, At);
  proj_gemm<<<dim3(EMBED / 64, MTOT / 256), 256, 0, stream>>>(At, WpT, Bp, out);
}
